// EnocoderBlock_48335561949887
// MI455X (gfx1250) — matrix operand order repaired and checked by reading, not compiled, not run on hardware
//
#include <hip/hip_runtime.h>

// ---------------------------------------------------------------------------
// Transformer encoder block for MI455X (gfx1250), bf16 WMMA compute path with
// Tensor-Data-Mover (TDM) double-buffered LDS staging in the GEMMs.
//   B=2, S=2048, D=1024, H=16, DFF=4096, DK=64
// ---------------------------------------------------------------------------

#define Bc   2
#define Sc   2048
#define Dc   1024
#define Hc   16
#define DFFc 4096
#define DKc  64
#define MRc  (Bc * Sc)          // 4096 token rows
#define EPSc 1e-6f
#define WPB  8                  // waves per block (256 threads)

typedef __attribute__((ext_vector_type(16))) __bf16 v16bf;
typedef __attribute__((ext_vector_type(8)))  __bf16 v8bf;
typedef __attribute__((ext_vector_type(8)))  float  v8f;
typedef unsigned int u32x4 __attribute__((ext_vector_type(4)));
typedef int          i32x8 __attribute__((ext_vector_type(8)));
typedef int          i32x4 __attribute__((ext_vector_type(4)));

static __device__ inline v8f wmma_bf16(v16bf a, v16bf b, v8f c) {
  // D = A(16x32, bf16) * B(32x16, bf16) + C(16x16, f32)
  return __builtin_amdgcn_wmma_f32_16x16x32_bf16(
      /*neg_a=*/false, a, /*neg_b=*/false, b,
      /*c_mod=*/(short)0, c, /*reuse_a=*/false, /*reuse_b=*/false);
}

// Load 16 contiguous bf16 (32B-aligned) as a B-fragment lane payload.
static __device__ inline v16bf load16(const __bf16* p) {
  return *reinterpret_cast<const v16bf*>(p);
}
// Build an A-fragment lane payload from two 8-element runs: [p, p+8) and [p+8, p+16).
static __device__ inline v16bf load8x2(const __bf16* p) {
  v8bf lo = *reinterpret_cast<const v8bf*>(p);
  v8bf hi = *reinterpret_cast<const v8bf*>(p + 8);
  v16bf r;
#pragma unroll
  for (int i = 0; i < 8; ++i) { r[i] = lo[i]; r[i + 8] = hi[i]; }
  return r;
}

// ---------------------------------------------------------------------------
// TDM: issue a 2-D tile load (Global -> LDS) via the Tensor Data Mover.
//   tile = tile_d1 rows x tile_d0 elements (bf16, contiguous rows in LDS),
//   source rows strided by `stride_el` elements.
// Descriptor bitfields per CDNA5 ISA sec 8.3/8.4 (D# group 0 + group 1).
// ---------------------------------------------------------------------------
static __device__ inline void tdm_load_2d(unsigned lds_off, const void* gaddr,
                                          unsigned tile_d0, unsigned tile_d1,
                                          unsigned tensor_d0, unsigned tensor_d1,
                                          unsigned long long stride_el) {
  const unsigned long long ga = (unsigned long long)gaddr;
  u32x4 g0;
  g0[0] = 1u;                                            // count=1 valid descriptor
  g0[1] = lds_off;                                       // lds_addr (bytes)
  g0[2] = (unsigned)(ga & 0xFFFFFFFFu);                  // global_addr[31:0]
  g0[3] = (unsigned)((ga >> 32) & 0x1FFFFFFu) | (2u << 30);  // addr[56:32] | type=2
  i32x8 g1;
  g1[0] = (int)(1u << 16);                               // wg_mask=0, data_size=1 (2B)
  g1[1] = (int)((tensor_d0 & 0xFFFFu) << 16);            // tensor_dim0[15:0]
  g1[2] = (int)(((tensor_d0 >> 16) & 0xFFFFu) | ((tensor_d1 & 0xFFFFu) << 16));
  g1[3] = (int)(((tensor_d1 >> 16) & 0xFFFFu) | (tile_d0 << 16));  // tile_dim0
  g1[4] = (int)(tile_d1 & 0xFFFFu);                      // tile_dim1, tile_dim2=0
  g1[5] = (int)(stride_el & 0xFFFFFFFFull);              // tensor_dim0_stride[31:0]
  g1[6] = (int)((stride_el >> 32) & 0xFFFFull);          // stride[47:32], dim1_stride=0
  g1[7] = 0;
  const i32x4 z4 = {0, 0, 0, 0};
#if __clang_major__ >= 23
  const i32x8 z8 = {0, 0, 0, 0, 0, 0, 0, 0};
  __builtin_amdgcn_tensor_load_to_lds(g0, g1, z4, z4, z8, 0);
#else
  __builtin_amdgcn_tensor_load_to_lds(g0, g1, z4, z4, 0);
#endif
}

static __device__ inline unsigned lds_offset_of(const void* p) {
  // Generic pointers to LDS carry the LDS byte offset in the low 32 bits.
  return (unsigned)(unsigned long long)p;
}

// ---------------------------------------------------------------------------
// fp32 -> bf16 cast
// ---------------------------------------------------------------------------
__global__ __launch_bounds__(256) void cast_bf16_kernel(const float* __restrict__ in,
                                                        __bf16* __restrict__ out,
                                                        size_t n) {
  size_t i = (size_t)blockIdx.x * blockDim.x + threadIdx.x;
  if (i < n) out[i] = (__bf16)in[i];
}

// ---------------------------------------------------------------------------
// V transpose: Vm[B*S, D] (head-major cols) -> Vt[B, H, DK, S]
// ---------------------------------------------------------------------------
__global__ __launch_bounds__(256) void transpose_v_kernel(const __bf16* __restrict__ Vm,
                                                          __bf16* __restrict__ Vt) {
  size_t i = (size_t)blockIdx.x * blockDim.x + threadIdx.x;
  if (i >= (size_t)MRc * Dc) return;
  int d = (int)(i % Dc);
  int row = (int)(i / Dc);
  int b = row / Sc, s = row % Sc;
  int h = d / DKc, dk = d % DKc;
  Vt[(((size_t)(b * Hc + h) * DKc) + dk) * Sc + s] = Vm[i];
}

// ---------------------------------------------------------------------------
// WMMA GEMM with TDM-staged LDS tiles: out[M,N] = act(A[M,K] * W[N,K]^T + b[N])
// Block tile 128x64 (8 waves of 32x32, arranged 4x2). Per 32-wide k-step,
// wave 0 DMAs the next A(128x32) and W(64x32) tiles to LDS (double buffered)
// while all waves run WMMAs on the current buffer. M%128==0, N%64==0, K%32==0.
// ---------------------------------------------------------------------------
template <typename OutT, bool RELU>
__global__ __launch_bounds__(256) void gemm_kernel(const __bf16* __restrict__ A,
                                                   const __bf16* __restrict__ W,
                                                   const float* __restrict__ bias,
                                                   OutT* __restrict__ out,
                                                   int Mv, int Nv, int Kv) {
  __shared__ alignas(64) __bf16 Ash[2][128][32];
  __shared__ alignas(64) __bf16 Wsh[2][64][32];

  const int lane = threadIdx.x & 31;
  const int wid  = threadIdx.x >> 5;
  const int nblk = Nv / 64;
  const int brow = blockIdx.x / nblk;
  const int bcol = blockIdx.x % nblk;

  const int mi = (wid & 3) * 32;       // wave row-tile inside 128-row block
  const int ni = (wid >> 2) * 32;      // wave col-tile inside 64-col block

  const int rA = lane & 15;            // A-frag row within 16-tile
  const int hA = (lane >> 4) * 16;     // A-frag K sub-offset
  const int cB = lane & 15;            // B-frag column within 16-tile
  const int hB = (lane >> 4) * 16;     // B-frag K sub-offset

  v8f z = {};
  v8f acc[2][2] = {{z, z}, {z, z}};

  const int ksteps = Kv / 32;
  const char* agbl = (const char*)(A + (size_t)brow * 128 * Kv);
  const char* wgbl = (const char*)(W + (size_t)bcol * 64 * Kv);

  if (wid == 0) {  // prologue: DMA first k-slab into buffer 0
    tdm_load_2d(lds_offset_of(&Ash[0][0][0]), agbl, 32, 128,
                (unsigned)Kv, (unsigned)Mv, (unsigned)Kv);
    tdm_load_2d(lds_offset_of(&Wsh[0][0][0]), wgbl, 32, 64,
                (unsigned)Kv, (unsigned)Nv, (unsigned)Kv);
  }

  for (int s = 0; s < ksteps; ++s) {
    const int buf = s & 1;
    if (wid == 0) __builtin_amdgcn_s_wait_tensorcnt(0);  // slab s landed
    __syncthreads();                                     // visible to all waves
    if (wid == 0 && (s + 1) < ksteps) {                  // overlap slab s+1 DMA
      const int nb = (s + 1) & 1;
      const size_t koff = (size_t)(s + 1) * 64;          // 32 bf16 = 64 bytes
      tdm_load_2d(lds_offset_of(&Ash[nb][0][0]), agbl + koff, 32, 128,
                  (unsigned)Kv, (unsigned)Mv, (unsigned)Kv);
      tdm_load_2d(lds_offset_of(&Wsh[nb][0][0]), wgbl + koff, 32, 64,
                  (unsigned)Kv, (unsigned)Nv, (unsigned)Kv);
    }
    v16bf af0 = load8x2(&Ash[buf][mi + rA][hA]);
    v16bf af1 = load8x2(&Ash[buf][mi + 16 + rA][hA]);
    v16bf bf0 = load16(&Wsh[buf][ni + cB][hB]);
    v16bf bf1 = load16(&Wsh[buf][ni + 16 + cB][hB]);
    acc[0][0] = wmma_bf16(af0, bf0, acc[0][0]);
    acc[0][1] = wmma_bf16(af0, bf1, acc[0][1]);
    acc[1][0] = wmma_bf16(af1, bf0, acc[1][0]);
    acc[1][1] = wmma_bf16(af1, bf1, acc[1][1]);
  }

  const int m0 = brow * 128 + mi;
  const int n0 = bcol * 64 + ni;
#pragma unroll
  for (int ti = 0; ti < 2; ++ti)
#pragma unroll
    for (int tj = 0; tj < 2; ++tj) {
      const int col = n0 + tj * 16 + (lane & 15);
      const float bs = bias[col];
#pragma unroll
      for (int v = 0; v < 8; ++v) {
        const int row = m0 + ti * 16 + ((lane >> 4) * 8) + v;
        float val = acc[ti][tj][v] + bs;
        if (RELU) val = fmaxf(val, 0.0f);
        out[(size_t)row * Nv + col] = (OutT)val;
      }
    }
}

// ---------------------------------------------------------------------------
// Flash attention: one wave per (b, h, 16-row q-tile); streams keys 32 at a
// time; QK^T and P*V via bf16 WMMA; online softmax with shfl reductions;
// P transposed C-layout -> A-layout through per-wave LDS tile.
// ---------------------------------------------------------------------------
__global__ __launch_bounds__(256) void flash_attn_kernel(const __bf16* __restrict__ Q,
                                                         const __bf16* __restrict__ K,
                                                         const __bf16* __restrict__ Vt,
                                                         const float* __restrict__ mask,
                                                         __bf16* __restrict__ ctx) {
  const int lane = threadIdx.x & 31;
  const int wid  = threadIdx.x >> 5;
  const int gw   = blockIdx.x * WPB + wid;
  const int QT   = Sc / 16;
  const int qt   = gw % QT;
  const int h    = (gw / QT) % Hc;
  const int b    = gw / (QT * Hc);
  const int qbase = qt * 16;

  __shared__ alignas(32) __bf16 Pl[WPB][16][32];

  const int rA = lane & 15;
  const int hA = (lane >> 4) * 16;
  const int cB = lane & 15;
  const int hB = (lane >> 4) * 16;
  const int rowoff = (lane >> 4) * 8;   // C-layout row offset for this half

  // Q fragments: 16 rows x 64 features -> two K=32 A-frags (resident in VGPRs).
  const __bf16* qrow = Q + (size_t)(b * Sc + qbase + rA) * Dc + h * DKc + hA;
  v16bf qf0 = load8x2(qrow);
  v16bf qf1 = load8x2(qrow + 32);

  float mrow[8], lrow[8];
  v8f z = {};
  v8f acc[4] = {z, z, z, z};
#pragma unroll
  for (int v = 0; v < 8; ++v) { mrow[v] = -1e30f; lrow[v] = 0.0f; }

  const float scl = 0.125f;  // 1/sqrt(DK)
  const __bf16* kbase_p = K + (size_t)(b * Sc) * Dc + h * DKc;
  const __bf16* vbase_p = Vt + (size_t)(b * Hc + h) * DKc * Sc;

  for (int kb = 0; kb < Sc; kb += 32) {
    __syncthreads();   // protect LDS P tile reuse across iterations

    // ---- scores: two 16x16 tiles (keys kb..kb+15, kb+16..kb+31) ----
    v8f s[2];
#pragma unroll
    for (int nt = 0; nt < 2; ++nt) {
      const __bf16* kp = kbase_p + (size_t)(kb + nt * 16 + cB) * Dc + hB;
      v8f c = {};
      c = wmma_bf16(qf0, load16(kp), c);
      c = wmma_bf16(qf1, load16(kp + 32), c);
      s[nt] = c;
    }

    // ---- online softmax update ----
    float corr[8];
#pragma unroll
    for (int v = 0; v < 8; ++v) {
      const int q = qbase + rowoff + v;
      float s0 = s[0][v] * scl + mask[(size_t)q * Sc + kb + (lane & 15)];
      float s1 = s[1][v] * scl + mask[(size_t)q * Sc + kb + 16 + (lane & 15)];
      float tm = fmaxf(s0, s1);
#pragma unroll
      for (int off = 8; off; off >>= 1) tm = fmaxf(tm, __shfl_xor(tm, off, 16));
      const float nm = fmaxf(mrow[v], tm);
      corr[v] = __expf(mrow[v] - nm);
      mrow[v] = nm;
      const float p0 = __expf(s0 - nm);
      const float p1 = __expf(s1 - nm);
      float rs = p0 + p1;
#pragma unroll
      for (int off = 8; off; off >>= 1) rs += __shfl_xor(rs, off, 16);
      lrow[v] = lrow[v] * corr[v] + rs;
      const int r = rowoff + v;
      Pl[wid][r][lane & 15]        = (__bf16)p0;   // C-layout -> LDS row-major
      Pl[wid][r][16 + (lane & 15)] = (__bf16)p1;
    }
#pragma unroll
    for (int t = 0; t < 4; ++t)
#pragma unroll
      for (int v = 0; v < 8; ++v) acc[t][v] *= corr[v];

    __syncthreads();   // P tile visible across the wave

    // ---- P(16x32) x V(32x64) ----
    v16bf pf = load8x2(&Pl[wid][rA][hA]);
#pragma unroll
    for (int t = 0; t < 4; ++t) {
      const __bf16* vp = vbase_p + (size_t)(t * 16 + cB) * Sc + kb + hB;
      acc[t] = wmma_bf16(pf, load16(vp), acc[t]);
    }
  }

  // ---- normalize and write ctx (bf16, [B,S,D] head-major cols) ----
#pragma unroll
  for (int t = 0; t < 4; ++t)
#pragma unroll
    for (int v = 0; v < 8; ++v) {
      const int r = rowoff + v;
      const float o = acc[t][v] / lrow[v];
      ctx[(size_t)(b * Sc + qbase + r) * Dc + h * DKc + t * 16 + (lane & 15)] = (__bf16)o;
    }
}

// ---------------------------------------------------------------------------
// Residual add + LayerNorm over D=1024. One block (256 threads) per row.
// Optionally also emits bf16 copy for the next GEMM.
// ---------------------------------------------------------------------------
template <bool WRITE_BF16>
__global__ __launch_bounds__(256) void add_ln_kernel(const float* __restrict__ a,
                                                     const float* __restrict__ r,
                                                     const float* __restrict__ alpha,
                                                     const float* __restrict__ gamma,
                                                     float* __restrict__ outf,
                                                     __bf16* __restrict__ outb) {
  const int row = blockIdx.x;
  const int tid = threadIdx.x;
  __shared__ float sbuf[256];

  const float* pa = a + (size_t)row * Dc;
  const float* pr = r + (size_t)row * Dc;
  float vloc[Dc / 256];
  float sum = 0.0f;
#pragma unroll
  for (int i = 0; i < Dc / 256; ++i) {
    const float v = pa[tid + i * 256] + pr[tid + i * 256];
    vloc[i] = v;
    sum += v;
  }
  sbuf[tid] = sum;
  __syncthreads();
  for (int s = 128; s; s >>= 1) { if (tid < s) sbuf[tid] += sbuf[tid + s]; __syncthreads(); }
  const float mean = sbuf[0] * (1.0f / Dc);
  __syncthreads();

  float vs = 0.0f;
#pragma unroll
  for (int i = 0; i < Dc / 256; ++i) {
    const float d = vloc[i] - mean;
    vs += d * d;
  }
  sbuf[tid] = vs;
  __syncthreads();
  for (int s = 128; s; s >>= 1) { if (tid < s) sbuf[tid] += sbuf[tid + s]; __syncthreads(); }
  const float var = sbuf[0] * (1.0f / Dc);
  const float inv = rsqrtf(var + EPSc);
  const float al = alpha[0], gm = gamma[0];

#pragma unroll
  for (int i = 0; i < Dc / 256; ++i) {
    const float o = al * (vloc[i] - mean) * inv + gm;
    outf[(size_t)row * Dc + tid + i * 256] = o;
    if (WRITE_BF16) outb[(size_t)row * Dc + tid + i * 256] = (__bf16)o;
  }
}

// ---------------------------------------------------------------------------
// Launcher
// ---------------------------------------------------------------------------
extern "C" void kernel_launch(void* const* d_in, const int* in_sizes, int n_in,
                              void* d_out, int out_size, void* d_ws, size_t ws_size,
                              hipStream_t stream) {
  const float* x    = (const float*)d_in[0];
  const float* mask = (const float*)d_in[1];
  const float* wq   = (const float*)d_in[2];
  const float* bq   = (const float*)d_in[3];
  const float* wk   = (const float*)d_in[4];
  const float* bk   = (const float*)d_in[5];
  const float* wv   = (const float*)d_in[6];
  const float* bv   = (const float*)d_in[7];
  const float* wo   = (const float*)d_in[8];
  const float* bo   = (const float*)d_in[9];
  const float* w1   = (const float*)d_in[10];
  const float* b1   = (const float*)d_in[11];
  const float* w2   = (const float*)d_in[12];
  const float* b2   = (const float*)d_in[13];
  const float* alpha= (const float*)d_in[14];
  const float* gamma= (const float*)d_in[15];
  float* out = (float*)d_out;

  char* p = (char*)d_ws;
  auto alloc = [&](size_t bytes) -> char* {
    char* r = p;
    p += (bytes + 255) & ~(size_t)255;
    return r;
  };

  __bf16* xb    = (__bf16*)alloc((size_t)MRc * Dc * 2);
  __bf16* wqb   = (__bf16*)alloc((size_t)Dc * Dc * 2);
  __bf16* wkb   = (__bf16*)alloc((size_t)Dc * Dc * 2);
  __bf16* wvb   = (__bf16*)alloc((size_t)Dc * Dc * 2);
  __bf16* wob   = (__bf16*)alloc((size_t)Dc * Dc * 2);
  __bf16* w1b   = (__bf16*)alloc((size_t)DFFc * Dc * 2);
  __bf16* w2b   = (__bf16*)alloc((size_t)Dc * DFFc * 2);
  __bf16* Qb    = (__bf16*)alloc((size_t)MRc * Dc * 2);
  __bf16* Kb    = (__bf16*)alloc((size_t)MRc * Dc * 2);
  __bf16* Vb    = (__bf16*)alloc((size_t)MRc * Dc * 2);
  __bf16* Vt    = (__bf16*)alloc((size_t)MRc * Dc * 2);
  __bf16* ctxb  = (__bf16*)alloc((size_t)MRc * Dc * 2);
  float*  attn  = (float*)alloc((size_t)MRc * Dc * 4);
  float*  out1  = (float*)alloc((size_t)MRc * Dc * 4);
  __bf16* out1b = (__bf16*)alloc((size_t)MRc * Dc * 2);
  __bf16* hid   = (__bf16*)alloc((size_t)MRc * DFFc * 2);
  float*  ffn   = (float*)alloc((size_t)MRc * Dc * 4);
  (void)ws_size; (void)in_sizes; (void)n_in; (void)out_size;

  auto cast = [&](const float* src, __bf16* dst, size_t n) {
    cast_bf16_kernel<<<(unsigned)((n + 255) / 256), 256, 0, stream>>>(src, dst, n);
  };
  cast(x,  xb,  (size_t)MRc * Dc);
  cast(wq, wqb, (size_t)Dc * Dc);
  cast(wk, wkb, (size_t)Dc * Dc);
  cast(wv, wvb, (size_t)Dc * Dc);
  cast(wo, wob, (size_t)Dc * Dc);
  cast(w1, w1b, (size_t)DFFc * Dc);
  cast(w2, w2b, (size_t)Dc * DFFc);

  auto gemm_blocks = [](int Mv, int Nv) -> unsigned {
    return (unsigned)((size_t)(Mv / 128) * (Nv / 64));
  };

  // Q/K/V projections -> bf16
  gemm_kernel<__bf16, false><<<gemm_blocks(MRc, Dc), 256, 0, stream>>>(
      xb, wqb, bq, Qb, MRc, Dc, Dc);
  gemm_kernel<__bf16, false><<<gemm_blocks(MRc, Dc), 256, 0, stream>>>(
      xb, wkb, bk, Kb, MRc, Dc, Dc);
  gemm_kernel<__bf16, false><<<gemm_blocks(MRc, Dc), 256, 0, stream>>>(
      xb, wvb, bv, Vb, MRc, Dc, Dc);

  // V -> [B,H,DK,S] for contiguous P*V B-fragments
  transpose_v_kernel<<<(unsigned)(((size_t)MRc * Dc + 255) / 256), 256, 0, stream>>>(Vb, Vt);

  // Flash attention -> ctx (bf16)
  flash_attn_kernel<<<(unsigned)((size_t)Bc * Hc * (Sc / 16) / WPB), 256, 0, stream>>>(
      Qb, Kb, Vt, mask, ctxb);

  // Output projection -> f32
  gemm_kernel<float, false><<<gemm_blocks(MRc, Dc), 256, 0, stream>>>(
      ctxb, wob, bo, attn, MRc, Dc, Dc);

  // out1 = LN(x + attn), plus bf16 copy for FFN
  add_ln_kernel<true><<<MRc, 256, 0, stream>>>(x, attn, alpha, gamma, out1, out1b);

  // FFN: hid = relu(out1 @ w1^T + b1) in bf16
  gemm_kernel<__bf16, true><<<gemm_blocks(MRc, DFFc), 256, 0, stream>>>(
      out1b, w1b, b1, hid, MRc, DFFc, Dc);
  // ffn = hid @ w2^T + b2 in f32
  gemm_kernel<float, false><<<gemm_blocks(MRc, Dc), 256, 0, stream>>>(
      hid, w2b, b2, ffn, MRc, Dc, DFFc);

  // out2 = LN(out1 + ffn) -> d_out
  add_ln_kernel<false><<<MRc, 256, 0, stream>>>(out1, ffn, alpha, gamma, out, nullptr);
}
